// Attention_1726576853699
// MI455X (gfx1250) — matrix operand order repaired and checked by reading, not compiled, not run on hardware
//
#include <hip/hip_runtime.h>

// Problem dims (fixed by the reference)
#define BH   16          // B*H
#define T    2048
#define NN   512         // head dim for Q (K of GEMM1)
#define DV   128
#define TWO_PI 6.28318530717958647692f

typedef __attribute__((ext_vector_type(16))) _Float16 v16h;
typedef __attribute__((ext_vector_type(8)))  float    v8f;
typedef __attribute__((ext_vector_type(4)))  unsigned int u32x4;
typedef __attribute__((ext_vector_type(8)))  int      i32x8;
typedef __attribute__((ext_vector_type(4)))  int      i32x4;

union Frag {
    v16h v;
    uint4 q[2];
};

// ---------------------------------------------------------------------------
// Kernel 1: fused RoPE + f32->f16.  QRh layout: [bh][t][n], n contiguous.
// ---------------------------------------------------------------------------
__global__ void rope_f16_kernel(const float* __restrict__ Q,
                                const float* __restrict__ freqs,
                                _Float16* __restrict__ QRh) {
    size_t idx = (size_t)blockIdx.x * blockDim.x + threadIdx.x;   // pair index
    const size_t total = (size_t)BH * T * (NN / 2);
    if (idx >= total) return;
    int n2 = (int)(idx & (NN / 2 - 1));
    size_t row = idx >> 8;                        // bh*T + t
    int t = (int)(row & (T - 1));
    size_t base = row * NN + 2 * n2;

    float f = freqs[2 * n2];
    float p = (float)t * f;
    p = (p - floorf(p)) * TWO_PI;
    float s, c;
    __sincosf(p, &s, &c);

    float q0 = Q[base], q1 = Q[base + 1];
    QRh[base]     = (_Float16)(q0 * c - q1 * s);
    QRh[base + 1] = (_Float16)(q1 * c + q0 * s);
}

// ---------------------------------------------------------------------------
// Kernel 2: V (f32, [bh][t][dv]) -> V^T (f16, [bh][dv][t])
// ---------------------------------------------------------------------------
__global__ void vtrans_f16_kernel(const float* __restrict__ V,
                                  _Float16* __restrict__ VhT) {
    size_t idx = (size_t)blockIdx.x * blockDim.x + threadIdx.x;
    const size_t total = (size_t)BH * T * DV;
    if (idx >= total) return;
    int dv = (int)(idx & (DV - 1));
    size_t r = idx >> 7;                 // bh*T + t
    int t = (int)(r & (T - 1));
    size_t bh = r >> 11;
    VhT[(bh * DV + dv) * (size_t)T + t] = (_Float16)V[idx];
}

// ---------------------------------------------------------------------------
// TDM: async 2-D tile load global->LDS with LDS row padding.
// D# bitfields per CDNA5 ISA ch.8 (group0: count/lds/global/type,
// group1: data_size=2B, pad, tensor/tile dims, dim0 stride).
// pad_interval v -> pad every 2^(v+1) DWORDs; pad_amount a -> (a+1) DWORDs.
// Toolchain uses the 6-arg builtin: (g0, g1, g2, g3, g_extra, cpol).
// ---------------------------------------------------------------------------
__device__ __forceinline__ void tdm_load_2d(const _Float16* lds_dst,
                                            const _Float16* gsrc,
                                            unsigned tensor_d0, unsigned tensor_d1,
                                            unsigned tile_d0,   unsigned tile_d1,
                                            unsigned long long stride_d0,
                                            unsigned pad_interval, unsigned pad_amount) {
    unsigned long long ga = (unsigned long long)(uintptr_t)gsrc;
    u32x4 g0;
    g0.x = 1u;                                           // count=1 (valid), user mode
    g0.y = (unsigned)(uintptr_t)lds_dst;                 // LDS byte address
    g0.z = (unsigned)(ga & 0xFFFFFFFFu);                 // global_addr[31:0]
    g0.w = (unsigned)((ga >> 32) & 0x1FFFFFFu) | (2u << 30);   // [56:32] | type=2
    i32x8 g1;
    g1[0] = (int)((1u << 16)                  // data_size = 2 bytes
                | (1u << 20)                  // pad_enable
                | (pad_interval << 22)
                | (pad_amount << 25));
    g1[1] = (int)((tensor_d0 & 0xFFFFu) << 16);                       // dim0[15:0]
    g1[2] = (int)(((tensor_d0 >> 16) & 0xFFFFu) | ((tensor_d1 & 0xFFFFu) << 16));
    g1[3] = (int)(((tensor_d1 >> 16) & 0xFFFFu) | ((tile_d0 & 0xFFFFu) << 16));
    g1[4] = (int)(tile_d1 & 0xFFFFu);                                 // tile_dim1, tile_dim2=0
    g1[5] = (int)(unsigned)(stride_d0 & 0xFFFFFFFFull);               // dim0_stride lo
    g1[6] = (int)(unsigned)((stride_d0 >> 32) & 0xFFFFull);           // dim0_stride hi
    g1[7] = 0;                                                        // dim1_stride (2-D)
    i32x4 g2 = {0, 0, 0, 0};
    i32x4 g3 = {0, 0, 0, 0};
    i32x8 g4 = {0, 0, 0, 0, 0, 0, 0, 0};
    __builtin_amdgcn_tensor_load_to_lds(g0, g1, g2, g3, g4, 0);
}

// ---------------------------------------------------------------------------
// LDS fragment loaders (16-bit WMMA operand layouts, wave32).
// ---------------------------------------------------------------------------
__device__ __forceinline__ v16h load_frag_a(const _Float16* base, int stride,
                                            int row, int kb, int g) {
    Frag f;
    const _Float16* p = base + row * stride + kb + 8 * g;
    f.q[0] = *(const uint4*)(p);
    f.q[1] = *(const uint4*)(p + 16);
    return f.v;
}

__device__ __forceinline__ v16h load_frag_b(const _Float16* base, int stride,
                                            int col, int kb, int g) {
    Frag f;
    const _Float16* p = base + col * stride + kb + 8 * g;
    f.q[0] = *(const uint4*)(p);
    f.q[1] = *(const uint4*)(p + 16);
    return f.v;
}

// ---------------------------------------------------------------------------
// Kernel 3: O = tril(QR QR^T, -1) @ V   per (bh, 128-row tile).
// 8 waves; wave w owns rows [16w,16w+16).  TDM double-buffers the K-chunks.
// LDS halves: qt0 qs0 qt1 qs1 (128x72 each), sc (128x136), vt (128x136)
// ---------------------------------------------------------------------------
#define QT_STRIDE 72
#define SC_STRIDE 136
#define QT0_OFF 0
#define QS0_OFF (128 * QT_STRIDE)
#define QT1_OFF (2 * 128 * QT_STRIDE)
#define QS1_OFF (3 * 128 * QT_STRIDE)
#define SC_OFF  (4 * 128 * QT_STRIDE)
#define VT_OFF  (SC_OFF + 128 * SC_STRIDE)
#define SMEM_HALVES (VT_OFF + 128 * SC_STRIDE)   // 71680 halves = 143360 B

__global__ __launch_bounds__(256, 1)
void attn_wmma_kernel(const _Float16* __restrict__ QRh,
                      const _Float16* __restrict__ VhT,
                      float* __restrict__ Out) {
    extern __shared__ _Float16 sm[];
    _Float16* qt[2] = { sm + QT0_OFF, sm + QT1_OFF };
    _Float16* qs[2] = { sm + QS0_OFF, sm + QS1_OFF };
    _Float16* sc = sm + SC_OFF;
    _Float16* vt = sm + VT_OFF;

    const int tid = threadIdx.x;
    const int w   = tid >> 5;
    const int ln  = tid & 31;
    const int g   = ln >> 4;
    const int l16 = ln & 15;

    const int ttile = blockIdx.x;
    const int bh    = blockIdx.y;
    const int t0    = ttile * 128;

    const _Float16* Qbh = QRh + (size_t)bh * T * NN;
    const _Float16* Vbh = VhT + (size_t)bh * DV * T;

    v8f oacc[8];
#pragma unroll
    for (int i = 0; i < 8; ++i) oacc[i] = v8f{};

    for (int st = 0; st <= ttile; ++st) {
        const int s0 = st * 128;

        // V^T tile DMA streams in underneath the whole GEMM1 K-loop (wave 2).
        if (w == 2)
            tdm_load_2d(vt, Vbh + s0, T, DV, 128, 128, T, /*pad*/5, 3);

        v8f sacc[8];
#pragma unroll
        for (int i = 0; i < 8; ++i) sacc[i] = v8f{};

        // Prologue: DMA chunk 0 into buffer 0 (wave0: rows t0.., wave1: rows s0..)
        if (w == 0)
            tdm_load_2d(qt[0], Qbh + (size_t)t0 * NN, NN, T, 64, 128, NN, /*pad*/4, 3);
        if (w == 1)
            tdm_load_2d(qs[0], Qbh + (size_t)s0 * NN, NN, T, 64, 128, NN, /*pad*/4, 3);

        // ---- GEMM1: scores[128x128] += QR_t[128x64] * QR_s[128x64]^T ----
#pragma unroll 2
        for (int i = 0; i < 8; ++i) {
            const int b  = i & 1;
            const int kc1 = (i + 1) * 64;
            if (w == 0 && i < 7)
                tdm_load_2d(qt[b ^ 1], Qbh + (size_t)t0 * NN + kc1, NN, T, 64, 128, NN, 4, 3);
            if (w == 1 && i < 7)
                tdm_load_2d(qs[b ^ 1], Qbh + (size_t)s0 * NN + kc1, NN, T, 64, 128, NN, 4, 3);
            if (w < 2) {
                if (i < 7) __builtin_amdgcn_s_wait_tensorcnt(1);  // chunk i done (in-order)
                else       __builtin_amdgcn_s_wait_tensorcnt(0);
            }
            __syncthreads();
            const _Float16* qtb = qt[b];
            const _Float16* qsb = qs[b];
#pragma unroll
            for (int ks = 0; ks < 64; ks += 32) {
                v16h a = load_frag_a(qtb, QT_STRIDE, 16 * w + l16, ks, g);
#pragma unroll
                for (int nt = 0; nt < 8; ++nt) {
                    v16h bb = load_frag_b(qsb, QT_STRIDE, 16 * nt + l16, ks, g);
                    sacc[nt] = __builtin_amdgcn_wmma_f32_16x16x32_f16(
                        false, a, false, bb, (short)0, sacc[nt], false, false);
                }
            }
            __syncthreads();
        }

        // ---- causal mask (strict lower) + spill scores to LDS as f16 ----
        const bool diag = (st == ttile);
#pragma unroll
        for (int nt = 0; nt < 8; ++nt) {
#pragma unroll
            for (int r = 0; r < 8; ++r) {
                int row = 16 * w + r + 8 * g;   // local t
                int col = 16 * nt + l16;        // local s
                float val = sacc[nt][r];
                if (diag && col >= row) val = 0.0f;
                sc[row * SC_STRIDE + col] = (_Float16)val;
            }
        }
        if (w == 2) __builtin_amdgcn_s_wait_tensorcnt(0);   // V^T tile in LDS
        __syncthreads();

        // ---- GEMM2: O[128x128] += scores[128x128] * V[128x128] ----
#pragma unroll
        for (int kk = 0; kk < 128; kk += 32) {
            v16h a = load_frag_a(sc, SC_STRIDE, 16 * w + l16, kk, g);
#pragma unroll
            for (int dt = 0; dt < 8; ++dt) {
                v16h bb = load_frag_b(vt, SC_STRIDE, 16 * dt + l16, kk, g);
                oacc[dt] = __builtin_amdgcn_wmma_f32_16x16x32_f16(
                    false, a, false, bb, (short)0, oacc[dt], false, false);
            }
        }
        __syncthreads();
    }

    // ---- store O (f32), C-layout: VGPR r -> row r+8g, lane%16 -> col ----
    float* Obh = Out + (size_t)bh * T * DV;
#pragma unroll
    for (int dt = 0; dt < 8; ++dt) {
#pragma unroll
        for (int r = 0; r < 8; ++r) {
            int row = t0 + 16 * w + r + 8 * g;
            Obh[(size_t)row * DV + 16 * dt + l16] = oacc[dt][r];
        }
    }
}

// ---------------------------------------------------------------------------
extern "C" void kernel_launch(void* const* d_in, const int* in_sizes, int n_in,
                              void* d_out, int out_size, void* d_ws, size_t ws_size,
                              hipStream_t stream) {
    const float* Q     = (const float*)d_in[0];
    const float* V     = (const float*)d_in[1];
    const float* freqs = (const float*)d_in[2];
    float* Out = (float*)d_out;

    _Float16* QRh = (_Float16*)d_ws;                                          // 32 MiB
    _Float16* VhT = (_Float16*)((char*)d_ws + (size_t)BH * T * NN * sizeof(_Float16)); // 8 MiB

    (void)hipFuncSetAttribute((const void*)attn_wmma_kernel,
                              hipFuncAttributeMaxDynamicSharedMemorySize,
                              SMEM_HALVES * (int)sizeof(_Float16));

    {
        const size_t total = (size_t)BH * T * (NN / 2);
        rope_f16_kernel<<<dim3((unsigned)((total + 255) / 256)), dim3(256), 0, stream>>>(
            Q, freqs, QRh);
    }
    {
        const size_t total = (size_t)BH * T * DV;
        vtrans_f16_kernel<<<dim3((unsigned)((total + 255) / 256)), dim3(256), 0, stream>>>(
            V, VhT);
    }
    {
        dim3 grid(T / 128, BH);
        attn_wmma_kernel<<<grid, dim3(256), SMEM_HALVES * sizeof(_Float16), stream>>>(
            QRh, VhT, Out);
    }
}
